// ThermodynamicAttention_12859132084860
// MI455X (gfx1250) — matrix operand order repaired and checked by reading, not compiled, not run on hardware
//
#include <hip/hip_runtime.h>
#include <hip/hip_bf16.h>

typedef __bf16 bf16_t;
typedef __attribute__((ext_vector_type(16))) __bf16 v16bf;
typedef __attribute__((ext_vector_type(8)))  __bf16 v8bf;
typedef __attribute__((ext_vector_type(8)))  float  v8f;

#define WMMA_BF16(a, b, c) \
  __builtin_amdgcn_wmma_f32_16x16x32_bf16(false, (a), false, (b), (short)0, (c), false, false)

// ---- fragment loaders (contiguous-k layouts, wave32) -------------------------
// B fragment (32x16 bf16): lane L -> col N = L%16, rows K = (L>>4)*16 + 0..15,
// contiguous in a K-major (Bt) LDS row. Two 16B loads.
static __device__ inline v16bf lds_load16(const bf16_t* p) {
  v8bf lo = *(const v8bf*)p;
  v8bf hi = *(const v8bf*)(p + 8);
  return __builtin_shufflevector(lo, hi, 0,1,2,3,4,5,6,7,8,9,10,11,12,13,14,15);
}
// A fragment (16x32 bf16): lane L -> row M = L%16, K elems (L>>4)*16+0..15.
// rowPtr points at (row, k=0) of the 32-wide slab.
static __device__ inline v16bf lds_load_afrag(const bf16_t* rowPtr, int lane) {
  int h8 = ((lane >> 4) & 1) * 16;
  v8bf lo = *(const v8bf*)(rowPtr + h8);
  v8bf hi = *(const v8bf*)(rowPtr + 8 + h8);
  return __builtin_shufflevector(lo, hi, 0,1,2,3,4,5,6,7,8,9,10,11,12,13,14,15);
}

// ---- fp32 -> bf16 convert ---------------------------------------------------
__global__ void f32_to_bf16_kernel(const float* __restrict__ in,
                                   bf16_t* __restrict__ out, int n) {
  for (int i = blockIdx.x * blockDim.x + threadIdx.x; i < n;
       i += gridDim.x * blockDim.x)
    out[i] = (bf16_t)in[i];
}

// ---- GEMM: C[m,n] = sum_k A[m,k]*W[n,k] + bias[n]  (torch Linear) ----------
// 256 thr = 8 waves (4 M-groups x 2 N-groups). Block tile 128x128, K-step 64.
// Wave tile 32(M) x 64(N): 8 f32 accum frags, 16 WMMA per K-step.
__global__ __launch_bounds__(256)
void gemm_bias_wmma(const bf16_t* __restrict__ A, const bf16_t* __restrict__ W,
                    const float* __restrict__ bias, void* __restrict__ Cout,
                    int K, int N, int out_f32) {
  __shared__ __align__(16) bf16_t sA[128 * 72];  // A tile rows padded to 72
  __shared__ __align__(16) bf16_t sW[128 * 72];  // W rows are K-major (= Bt)
  const int tid = threadIdx.x, lane = tid & 31, wave = tid >> 5;
  const int hi = lane >> 4;
  const int m0 = blockIdx.y * 128, n0 = blockIdx.x * 128;
  const int mrow = (wave >> 1) * 32;   // wave's M offset inside block tile
  const int ncol = (wave & 1) * 64;    // wave's N offset inside block tile

  v8f acc[2][4];
  v8f zero = {0.f, 0.f, 0.f, 0.f, 0.f, 0.f, 0.f, 0.f};
#pragma unroll
  for (int mi = 0; mi < 2; ++mi)
#pragma unroll
    for (int nf = 0; nf < 4; ++nf) acc[mi][nf] = zero;

  for (int kk = 0; kk < K; kk += 64) {
#pragma unroll
    for (int j = 0; j < 8; ++j) {        // 128x64 bf16 each, uint2 = 4 elems
      int e = tid + j * 256;             // 0..2047
      int r = e >> 4, q = e & 15;
      *(uint2*)&sA[r * 72 + q * 4] =
          *(const uint2*)&A[(size_t)(m0 + r) * K + kk + q * 4];
      *(uint2*)&sW[r * 72 + q * 4] =
          *(const uint2*)&W[(size_t)(n0 + r) * K + kk + q * 4];
    }
    __syncthreads();
#pragma unroll
    for (int ks = 0; ks < 64; ks += 32) {
      v16bf aF[2];
#pragma unroll
      for (int mi = 0; mi < 2; ++mi)
        aF[mi] = lds_load_afrag(
            &sA[(mrow + mi * 16 + (lane & 15)) * 72 + ks], lane);
#pragma unroll
      for (int nf = 0; nf < 4; ++nf) {
        v16bf bF = lds_load16(
            &sW[(ncol + nf * 16 + (lane & 15)) * 72 + ks + (hi << 4)]);
#pragma unroll
        for (int mi = 0; mi < 2; ++mi)
          acc[mi][nf] = WMMA_BF16(aF[mi], bF, acc[mi][nf]);
      }
    }
    __syncthreads();
  }
#pragma unroll
  for (int nf = 0; nf < 4; ++nf) {
    int col = n0 + ncol + nf * 16 + (lane & 15);
    float bv = bias[col];
#pragma unroll
    for (int mi = 0; mi < 2; ++mi)
#pragma unroll
      for (int r = 0; r < 8; ++r) {
        int row = m0 + mrow + mi * 16 + r + 8 * hi;  // C: M = r + 8*(lane>>4)
        float v = acc[mi][nf][r] + bv;
        if (out_f32) ((float*)Cout)[(size_t)row * N + col] = v;
        else         ((bf16_t*)Cout)[(size_t)row * N + col] = (bf16_t)v;
      }
  }
}

// ---- thermodynamic attention (expectation of Bernoulli MC = sigmoid) -------
// grid: (S/64 query tiles, B*H). Block 128 thr (4 waves), each wave owns 16 q.
// 64-key chunks: acc += p_chunk @ V_chunk ; rowsum += sum(p); out = acc/rowsum.
__global__ __launch_bounds__(128)
void thermo_attn_wmma(const bf16_t* __restrict__ Qb, const bf16_t* __restrict__ Kb,
                      const bf16_t* __restrict__ Vb, bf16_t* __restrict__ Ob,
                      int S, int Dm, int H) {
  const int dk = 64;
  const float scale = 0.125f;  // 1/sqrt(64), BETA = 1
  __shared__ __align__(16) bf16_t sQ[64 * 72];
  __shared__ __align__(16) bf16_t sK[64 * 72];
  __shared__ __align__(16) bf16_t sVt[64 * 72];     // transposed: [d][key0..63]
  __shared__ __align__(16) bf16_t sP[4][16 * 72];   // per-wave sigmoid tile
  __shared__ float sRS[4 * 16];

  const int tid = threadIdx.x, lane = tid & 31, wave = tid >> 5;
  const int hi = lane >> 4;
  const int bh = blockIdx.y, b = bh / H, h = bh % H;
  const int q0 = blockIdx.x * 64;
  const bf16_t* Qh = Qb + (size_t)b * S * Dm + h * dk;
  const bf16_t* Kh = Kb + (size_t)b * S * Dm + h * dk;
  const bf16_t* Vh = Vb + (size_t)b * S * Dm + h * dk;
  bf16_t*       Oh = Ob + (size_t)b * S * Dm + h * dk;

#pragma unroll
  for (int j = 0; j < 8; ++j) {           // Q tile 64x64 resident
    int e = tid + j * 128;                // 0..1023
    int r = e >> 4, q = e & 15;
    *(uint2*)&sQ[r * 72 + q * 4] =
        *(const uint2*)&Qh[(size_t)(q0 + r) * Dm + q * 4];
  }

  v8f oacc[4];
  v8f zero = {0.f, 0.f, 0.f, 0.f, 0.f, 0.f, 0.f, 0.f};
#pragma unroll
  for (int i = 0; i < 4; ++i) oacc[i] = zero;
  float rowsum = 0.f;

  for (int kt = 0; kt < S; kt += 64) {
#pragma unroll
    for (int j = 0; j < 8; ++j) {         // K tile 64x64
      int e = tid + j * 128;
      int r = e >> 4, q = e & 15;
      *(uint2*)&sK[r * 72 + q * 4] =
          *(const uint2*)&Kh[(size_t)(kt + r) * Dm + q * 4];
    }
#pragma unroll
    for (int j = 0; j < 8; ++j) {         // V tile 64x64, scattered transposed
      int e = tid + j * 128;
      int key = e >> 4, q = e & 15;
      const bf16_t* src = &Vh[(size_t)(kt + key) * Dm + q * 4];
#pragma unroll
      for (int i = 0; i < 4; ++i) sVt[(q * 4 + i) * 72 + key] = src[i];
    }
    __syncthreads();

    v8f sc[4] = {zero, zero, zero, zero};  // scores 16q x 64k
#pragma unroll
    for (int d0 = 0; d0 < 64; d0 += 32) {
      v16bf aQ = lds_load_afrag(&sQ[(wave * 16 + (lane & 15)) * 72 + d0], lane);
#pragma unroll
      for (int nf = 0; nf < 4; ++nf) {
        // Bt[key][d] = sK row; contiguous 16 elems along d
        v16bf bK = lds_load16(&sK[(nf * 16 + (lane & 15)) * 72 + d0 + (hi << 4)]);
        sc[nf] = WMMA_BF16(aQ, bK, sc[nf]);
      }
    }
    // p = sigmoid(scale*score); stage bf16 P tile (C layout -> row major)
#pragma unroll
    for (int nf = 0; nf < 4; ++nf)
#pragma unroll
      for (int r = 0; r < 8; ++r) {
        float pv = 1.0f / (1.0f + __expf(-scale * sc[nf][r]));
        sP[wave][(r + 8 * hi) * 72 + nf * 16 + (lane & 15)] = (bf16_t)pv;
      }
    asm volatile("s_wait_dscnt 0" ::: "memory");
    if (lane < 16) {
      float s = 0.f;
      const bf16_t* pr = &sP[wave][lane * 72];
#pragma unroll
      for (int c = 0; c < 64; ++c) s += (float)pr[c];
      rowsum += s;
    }
    // acc += P(16x64) @ V(64x64), split over two 32-key slabs
#pragma unroll
    for (int kh = 0; kh < 64; kh += 32) {
      v16bf aP =
          lds_load_afrag(&sP[wave][(lane & 15) * 72 + kh], lane);
#pragma unroll
      for (int nf = 0; nf < 4; ++nf) {
        v16bf bV = lds_load16(
            &sVt[(nf * 16 + (lane & 15)) * 72 + kh + (hi << 4)]);
        oacc[nf] = WMMA_BF16(aP, bV, oacc[nf]);
      }
    }
    __syncthreads();
  }

  if (lane < 16) sRS[wave * 16 + lane] = rowsum;
  asm volatile("s_wait_dscnt 0" ::: "memory");
#pragma unroll
  for (int r = 0; r < 8; ++r) {
    int M = r + 8 * hi;
    float rs = sRS[wave * 16 + M];
    float inv = 1.0f / fmaxf(rs, 1e-20f);
#pragma unroll
    for (int nf = 0; nf < 4; ++nf) {
      int row = q0 + wave * 16 + M;
      int col = nf * 16 + (lane & 15);
      Oh[(size_t)row * Dm + col] = (bf16_t)(oacc[nf][r] * inv);
    }
  }
}

// ---- host side --------------------------------------------------------------
extern "C" void kernel_launch(void* const* d_in, const int* in_sizes, int n_in,
                              void* d_out, int out_size, void* d_ws, size_t ws_size,
                              hipStream_t stream) {
  (void)in_sizes; (void)n_in; (void)out_size; (void)ws_size;
  const int B = 2, S = 1024, D = 768, H = 12;
  const int M = B * S;  // 2048

  const float* x  = (const float*)d_in[0];
  const float* Wq = (const float*)d_in[1];
  const float* bq = (const float*)d_in[2];
  const float* Wk = (const float*)d_in[3];
  const float* bk = (const float*)d_in[4];
  const float* Wv = (const float*)d_in[5];
  const float* bv = (const float*)d_in[6];
  const float* Wo = (const float*)d_in[7];
  const float* bo = (const float*)d_in[8];

  char* ws = (char*)d_ws;
  size_t off = 0;
  auto take = [&](size_t bytes) {
    size_t o = off;
    off += (bytes + 255) & ~(size_t)255;
    return o;
  };
  const size_t xBytes = (size_t)M * D * sizeof(bf16_t);
  const size_t wBytes = (size_t)D * D * sizeof(bf16_t);
  bf16_t* xb    = (bf16_t*)(ws + take(xBytes));
  bf16_t* Wqb   = (bf16_t*)(ws + take(wBytes));
  bf16_t* Wkb   = (bf16_t*)(ws + take(wBytes));
  bf16_t* Wvb   = (bf16_t*)(ws + take(wBytes));
  bf16_t* Wob   = (bf16_t*)(ws + take(wBytes));
  bf16_t* Qbf   = (bf16_t*)(ws + take(xBytes));
  bf16_t* Kbf   = (bf16_t*)(ws + take(xBytes));
  bf16_t* Vbf   = (bf16_t*)(ws + take(xBytes));
  bf16_t* attnb = (bf16_t*)(ws + take(xBytes));

  // converts
  f32_to_bf16_kernel<<<2048, 256, 0, stream>>>(x, xb, M * D);
  f32_to_bf16_kernel<<<1024, 256, 0, stream>>>(Wq, Wqb, D * D);
  f32_to_bf16_kernel<<<1024, 256, 0, stream>>>(Wk, Wkb, D * D);
  f32_to_bf16_kernel<<<1024, 256, 0, stream>>>(Wv, Wvb, D * D);
  f32_to_bf16_kernel<<<1024, 256, 0, stream>>>(Wo, Wob, D * D);

  // QKV projections (bf16 out, head-major layout falls out of row layout)
  dim3 gg(D / 128, M / 128);  // (6, 16)
  gemm_bias_wmma<<<gg, 256, 0, stream>>>(xb, Wqb, bq, Qbf, D, D, 0);
  gemm_bias_wmma<<<gg, 256, 0, stream>>>(xb, Wkb, bk, Kbf, D, D, 0);
  gemm_bias_wmma<<<gg, 256, 0, stream>>>(xb, Wvb, bv, Vbf, D, D, 0);

  // attention
  thermo_attn_wmma<<<dim3(S / 64, B * H), 128, 0, stream>>>(Qbf, Kbf, Vbf, attnb,
                                                            S, D, H);
  // output projection (f32 out)
  gemm_bias_wmma<<<gg, 256, 0, stream>>>(attnb, Wob, bo, (float*)d_out, D, D, 1);
}
